// HybridAttention_37254546325711
// MI455X (gfx1250) — matrix operand order repaired and checked by reading, not compiled, not run on hardware
//
#include <hip/hip_runtime.h>
#include <hip/hip_bf16.h>

#define BATCH 2
#define SEQ   2048
#define MDIM  1024
#define NH    16
#define NKV   4
#define HD    64

typedef __attribute__((ext_vector_type(16))) _Float16 v16h;
typedef __attribute__((ext_vector_type(16))) __bf16   v16bf;
typedef __attribute__((ext_vector_type(8)))  float    v8f;

union HFrag { v16h v; unsigned int p[8]; unsigned short u[16]; };
union BFrag { v16bf v; unsigned int p[8]; unsigned short u[16]; };

static __device__ __forceinline__ unsigned short f32_to_bf16(float f) {
  union { float f; unsigned int u; } cv; cv.f = f;
  unsigned int r = cv.u + 0x7FFFu + ((cv.u >> 16) & 1u);
  return (unsigned short)(r >> 16);
}
static __device__ __forceinline__ unsigned short f32_to_f16b(float f) {
  union { _Float16 h; unsigned short u; } cv; cv.h = (_Float16)f; return cv.u;
}

// ---------------- weight fake-quant -> bf16 ----------------
__global__ __launch_bounds__(256) void k_quant(
    const float* __restrict__ w, unsigned short* __restrict__ out, int cols, int nbits)
{
  const int row = blockIdx.x, tid = threadIdx.x;
  const float* wr = w + (size_t)row * cols;
  float mx = 0.f;
  for (int c = tid; c < cols; c += 256) mx = fmaxf(mx, fabsf(wr[c]));
  __shared__ float red[256];
  red[tid] = mx;
  __syncthreads();
  for (int s = 128; s > 0; s >>= 1) { if (tid < s) red[tid] = fmaxf(red[tid], red[tid + s]); __syncthreads(); }
  const float wmax = fmaxf(red[0], 1e-5f);
  const float hv = (float)(1 << (nbits - 1));
  for (int c = tid; c < cols; c += 256) {
    float wi = rintf(wr[c] / wmax * hv);           // round-half-even like jnp.round
    wi = fminf(fmaxf(wi, -hv), hv - 1.f);
    out[(size_t)row * cols + c] = f32_to_bf16(wi / hv * wmax);
  }
}

__global__ __launch_bounds__(256) void k_cvt_bf16(
    const float* __restrict__ in, unsigned short* __restrict__ out, int n)
{
  int idx = blockIdx.x * 256 + threadIdx.x;
  if (idx < n) out[idx] = f32_to_bf16(in[idx]);
}

// ---------------- C[M,N] = A[M,K] * B[N,K]^T, bf16 WMMA ----------------
__global__ __launch_bounds__(256) void k_gemm_bf16(
    const unsigned short* __restrict__ A, const unsigned short* __restrict__ Bm,
    float* __restrict__ C, int M, int N, int K)
{
  (void)M;
  const int lane = threadIdx.x & 31, wave = threadIdx.x >> 5;
  const int hh = lane >> 4, ln = lane & 15;
  const int tn = blockIdx.x * 8 + wave;            // 16-wide N tile per wave
  const int tm = blockIdx.y;                       // 16-wide M tile per block-row
  const unsigned short* Arow = A + (size_t)(tm * 16 + ln) * K;
  const unsigned short* Brow = Bm + (size_t)(tn * 16 + ln) * K;
  v8f acc = {};
  for (int k0 = 0; k0 < K; k0 += 32) {
    if (k0 + 64 <= K) {                            // gfx1250 global_prefetch_b8
      __builtin_prefetch(Arow + k0 + 64, 0, 0);
      __builtin_prefetch(Brow + k0 + 64, 0, 0);
    }
    BFrag a, b;
#pragma unroll
    for (int j = 0; j < 8; ++j) {
      int ka = 16 * hh + 2 * j;  // A layout
      a.p[j] = *(const unsigned int*)(Arow + k0 + ka);
      b.p[j] = *(const unsigned int*)(Brow + k0 + 16 * hh + 2 * j);      // B layout
    }
    acc = __builtin_amdgcn_wmma_f32_16x16x32_bf16(false, a.v, false, b.v,
                                                  (short)0, acc, false, false);
  }
  float* Cp = C + (size_t)(tm * 16 + 8 * hh) * N + tn * 16 + ln;
#pragma unroll
  for (int i = 0; i < 8; ++i) Cp[(size_t)i * N] = acc[i];
}

// ---------------- rmsnorm + rope + gain -> f16 heads ----------------
__global__ __launch_bounds__(64) void k_qkv_prep(
    const float* __restrict__ q_raw, const float* __restrict__ k_raw,
    const float* __restrict__ v_raw, const float* __restrict__ q_gain,
    unsigned short* __restrict__ qf, unsigned short* __restrict__ kf,
    unsigned short* __restrict__ vt)
{
  const int bt = blockIdx.x;
  const int b = bt / SEQ, t = bt % SEQ;
  const int slot = blockIdx.y;                     // 0..15 q, 16..19 k, 20..23 v
  const int d = threadIdx.x;
  __shared__ float ssq[64];
  __shared__ float sval[64];

  if (slot < NH + NKV) {
    const bool isq = slot < NH;
    const int hq = isq ? slot : (slot - NH);
    const float* src = isq ? (q_raw + (size_t)bt * MDIM + hq * HD)
                           : (k_raw + (size_t)bt * (NKV * HD) + hq * HD);
    float x = src[d];
    ssq[d] = x * x;
    __syncthreads();
    for (int s = 32; s > 0; s >>= 1) { if (d < s) ssq[d] += ssq[d + s]; __syncthreads(); }
    float xn = x * rsqrtf(ssq[0] * (1.0f / 64.0f) + 1.1920929e-07f);
    sval[d] = xn;
    __syncthreads();
    float outv = xn;
    if (d < 32) {                                  // rope dims, half = 16
      const int i = d & 15;
      const float inv_freq = expf(-(float)i * (9.210340371976184f / 16.0f)); // 10000^(-2i/32)
      const float ang = (float)t * inv_freq;
      const float c = cosf(ang), s = sinf(ang);
      const float x1 = sval[i], x2 = sval[i + 16];
      outv = (d < 16) ? (x1 * c + x2 * s) : (-x1 * s + x2 * c);
    }
    if (isq) {
      outv *= q_gain[hq];
      qf[(((size_t)(b * NH + hq)) * SEQ + t) * HD + d] = f32_to_f16b(outv);
    } else {
      kf[(((size_t)(b * NKV + hq)) * SEQ + t) * HD + d] = f32_to_f16b(outv);
    }
  } else {
    const int hk = slot - (NH + NKV);
    const float v = v_raw[(size_t)bt * (NKV * HD) + hk * HD + d];
    vt[(((size_t)(b * NKV + hk)) * HD + d) * SEQ + t] = f32_to_f16b(v);   // V transposed
  }
}

// ---------------- flash attention, f16 WMMA ----------------
__global__ __launch_bounds__(256) void k_attn(
    const unsigned short* __restrict__ qf, const unsigned short* __restrict__ kfm,
    const unsigned short* __restrict__ vtm, float* __restrict__ y)
{
  const int lane = threadIdx.x & 31;
  const int wave = threadIdx.x >> 5;
  const int hh = lane >> 4;
  const int ln = lane & 15;
  const int bh = blockIdx.y;                        // b*NH + h
  const int b  = bh / NH;
  const int h  = bh % NH;
  const int kv = h >> 2;                            // GQA: 4 q heads per kv head
  const int qbase = blockIdx.x * 128 + wave * 16;   // 16 q rows per wave

  const unsigned short* Q  = qf  + ((size_t)bh * SEQ + qbase) * HD;
  const unsigned short* Kp = kfm + ((size_t)(b * NKV + kv) * SEQ) * HD;   // [T][64]
  const unsigned short* Vt = vtm + ((size_t)(b * NKV + kv) * HD) * SEQ;   // [64][T]

  HFrag qa[2];                                      // Q A-fragments, d in [0,32),[32,64)
#pragma unroll
  for (int kh = 0; kh < 2; ++kh)
#pragma unroll
    for (int j = 0; j < 8; ++j) {
      int koff = 16 * hh + 2 * j;
      qa[kh].p[j] = *(const unsigned int*)(Q + (size_t)ln * HD + kh * 32 + koff);
    }

  v8f o[4] = {};
  float mrow[8], lrow[8];
#pragma unroll
  for (int i = 0; i < 8; ++i) { mrow[i] = -__builtin_inff(); lrow[i] = 0.f; }

  __shared__ __align__(16) unsigned short pshare[8][16][32];  // per-wave P tile

  for (int kb = 0; kb < qbase + 16; kb += 32) {     // per-wave causal bound
    v8f s0 = {}, s1 = {};
#pragma unroll
    for (int kh = 0; kh < 2; ++kh) {
      HFrag kf0, kf1;
#pragma unroll
      for (int j = 0; j < 8; ++j) {
        int koff = kh * 32 + 16 * hh + 2 * j;
        kf0.p[j] = *(const unsigned int*)(Kp + (size_t)(kb + ln) * HD + koff);
        kf1.p[j] = *(const unsigned int*)(Kp + (size_t)(kb + 16 + ln) * HD + koff);
      }
      s0 = __builtin_amdgcn_wmma_f32_16x16x32_f16(false, qa[kh].v, false, kf0.v, (short)0, s0, false, false);
      s1 = __builtin_amdgcn_wmma_f32_16x16x32_f16(false, qa[kh].v, false, kf1.v, (short)0, s1, false, false);
    }
#pragma unroll
    for (int i = 0; i < 8; ++i) {                   // online softmax per row
      const int r = qbase + 8 * hh + i;
      float a0 = (kb + ln      <= r) ? s0[i] * 0.125f : -__builtin_inff();
      float a1 = (kb + 16 + ln <= r) ? s1[i] * 0.125f : -__builtin_inff();
      float mx = fmaxf(a0, a1);
#pragma unroll
      for (int m = 1; m < 16; m <<= 1) mx = fmaxf(mx, __shfl_xor(mx, m, 32));
      const float nm = fmaxf(mrow[i], mx);
      const float alpha = expf(mrow[i] - nm);
      const float p0 = expf(a0 - nm);
      const float p1 = expf(a1 - nm);
      float rs = p0 + p1;
#pragma unroll
      for (int m = 1; m < 16; m <<= 1) rs += __shfl_xor(rs, m, 32);
      lrow[i] = lrow[i] * alpha + rs;
      mrow[i] = nm;
      o[0][i] *= alpha; o[1][i] *= alpha; o[2][i] *= alpha; o[3][i] *= alpha;
      pshare[wave][8 * hh + i][ln]      = f32_to_f16b(p0);
      pshare[wave][8 * hh + i][16 + ln] = f32_to_f16b(p1);
    }
    HFrag pa;                                       // C-layout -> A-layout via wave-private LDS
#pragma unroll
    for (int j = 0; j < 8; ++j) {
      int koff = 16 * hh + 2 * j;
      pa.p[j] = *(const unsigned int*)&pshare[wave][ln][koff];
    }
#pragma unroll
    for (int dt = 0; dt < 4; ++dt) {                // O += P @ V, 4 d-tiles
      HFrag vb;
#pragma unroll
      for (int j = 0; j < 8; ++j) {
        int krow = kb + 16 * hh + 2 * j;
        vb.p[j] = *(const unsigned int*)(Vt + (size_t)(dt * 16 + ln) * SEQ + krow);
      }
      o[dt] = __builtin_amdgcn_wmma_f32_16x16x32_f16(false, pa.v, false, vb.v, (short)0, o[dt], false, false);
    }
  }

  float* Y = y + ((size_t)bh * SEQ + qbase) * HD;
#pragma unroll
  for (int i = 0; i < 8; ++i) {
    const float inv = 1.0f / lrow[i];
    const size_t r = (size_t)(8 * hh + i) * HD;
    Y[r + ln]      = o[0][i] * inv;
    Y[r + 16 + ln] = o[1][i] * inv;
    Y[r + 32 + ln] = o[2][i] * inv;
    Y[r + 48 + ln] = o[3][i] * inv;
  }
}

// ---------------- v-direction rejection + bf16 for final GEMM ----------------
__global__ __launch_bounds__(64) void k_vproj(
    const float* __restrict__ y, const float* __restrict__ v_raw,
    unsigned short* __restrict__ y_bf)
{
  const int bt = blockIdx.x, h = blockIdx.y, d = threadIdx.x;
  const int b = bt / SEQ, t = bt % SEQ;
  const int kv = h >> 2;
  const float yv = y[(((size_t)(b * NH + h)) * SEQ + t) * HD + d];
  const float vv = v_raw[(size_t)bt * (NKV * HD) + kv * HD + d];
  __shared__ float syv[64], svv[64];
  syv[d] = yv * vv; svv[d] = vv * vv;
  __syncthreads();
  for (int s = 32; s > 0; s >>= 1) { if (d < s) { syv[d] += syv[d + s]; svv[d] += svv[d + s]; } __syncthreads(); }
  const float denom = fmaxf(sqrtf(svv[0]), 1e-12f);
  const float coef = syv[0] / (denom * denom);
  y_bf[(size_t)bt * MDIM + h * HD + d] = f32_to_bf16(yv - coef * vv);
}

extern "C" void kernel_launch(void* const* d_in, const int* in_sizes, int n_in,
                              void* d_out, int out_size, void* d_ws, size_t ws_size,
                              hipStream_t stream)
{
  (void)in_sizes; (void)n_in; (void)out_size; (void)ws_size;
  const float* x      = (const float*)d_in[0];
  const float* w_q    = (const float*)d_in[1];
  const float* w_k    = (const float*)d_in[2];
  const float* w_v    = (const float*)d_in[3];
  const float* w_proj = (const float*)d_in[4];
  const float* q_gain = (const float*)d_in[5];

  char* ws = (char*)d_ws;
  size_t off = 0;
  auto alloc = [&](size_t bytes) -> char* {
    char* p = ws + off; off += (bytes + 255) & ~(size_t)255; return p;
  };
  unsigned short* wq_bf = (unsigned short*)alloc((size_t)MDIM * MDIM * 2);
  unsigned short* wk_bf = (unsigned short*)alloc((size_t)NKV * HD * MDIM * 2);
  unsigned short* wv_bf = (unsigned short*)alloc((size_t)NKV * HD * MDIM * 2);
  unsigned short* wp_bf = (unsigned short*)alloc((size_t)MDIM * MDIM * 2);
  unsigned short* x_bf  = (unsigned short*)alloc((size_t)BATCH * SEQ * MDIM * 2);
  float* q_raw = (float*)alloc((size_t)BATCH * SEQ * MDIM * 4);
  float* k_raw = (float*)alloc((size_t)BATCH * SEQ * NKV * HD * 4);
  float* v_raw = (float*)alloc((size_t)BATCH * SEQ * NKV * HD * 4);
  unsigned short* qf = (unsigned short*)alloc((size_t)BATCH * NH * SEQ * HD * 2);
  unsigned short* kf = (unsigned short*)alloc((size_t)BATCH * NKV * SEQ * HD * 2);
  unsigned short* vt = (unsigned short*)alloc((size_t)BATCH * NKV * HD * SEQ * 2);
  float* y_att = (float*)alloc((size_t)BATCH * NH * SEQ * HD * 4);
  unsigned short* y_bf = (unsigned short*)alloc((size_t)BATCH * SEQ * MDIM * 2);

  k_quant<<<MDIM, 256, 0, stream>>>(w_q, wq_bf, MDIM, 6);
  k_quant<<<NKV * HD, 256, 0, stream>>>(w_k, wk_bf, MDIM, 6);
  k_quant<<<NKV * HD, 256, 0, stream>>>(w_v, wv_bf, MDIM, 5);
  k_quant<<<MDIM, 256, 0, stream>>>(w_proj, wp_bf, MDIM, 5);

  const int nx = BATCH * SEQ * MDIM;
  k_cvt_bf16<<<nx / 256, 256, 0, stream>>>(x, x_bf, nx);

  const int Mrows = BATCH * SEQ;
  k_gemm_bf16<<<dim3(MDIM / 128, Mrows / 16), 256, 0, stream>>>(x_bf, wq_bf, q_raw, Mrows, MDIM, MDIM);
  k_gemm_bf16<<<dim3((NKV * HD) / 128, Mrows / 16), 256, 0, stream>>>(x_bf, wk_bf, k_raw, Mrows, NKV * HD, MDIM);
  k_gemm_bf16<<<dim3((NKV * HD) / 128, Mrows / 16), 256, 0, stream>>>(x_bf, wv_bf, v_raw, Mrows, NKV * HD, MDIM);

  k_qkv_prep<<<dim3(Mrows, NH + 2 * NKV), 64, 0, stream>>>(q_raw, k_raw, v_raw, q_gain, qf, kf, vt);

  k_attn<<<dim3(SEQ / 128, BATCH * NH), 256, 0, stream>>>(qf, kf, vt, y_att);

  k_vproj<<<dim3(Mrows, NH), 64, 0, stream>>>(y_att, v_raw, y_bf);

  k_gemm_bf16<<<dim3(MDIM / 128, Mrows / 16), 256, 0, stream>>>(y_bf, wp_bf, (float*)d_out, Mrows, MDIM, MDIM);
}
